// TV_26268019982520
// MI455X (gfx1250) — compile-verified, matrix operand order checked by reading, not run on hardware
//
#include <hip/hip_runtime.h>
#include <hip/hip_bf16.h>

typedef __attribute__((ext_vector_type(16))) _Float16 v16h;
typedef __attribute__((ext_vector_type(8)))  float    v8f;

namespace {
constexpr int Bc = 8, Cc = 32, Hc = 512, Wc = 512;
constexpr int HS = 32;              // rows per block strip
constexpr int WAVES_PER_BLOCK = 8;  // 256 threads = 8 wave32
constexpr int PIX_PER_WAVE = 16;    // one WMMA tile of output pixels per wave
}

// Total-variation with channel reduction done by V_WMMA_F32_16X16X32_F16:
//   B (32x16 f16) = tv[channel][pixel], A = ones(16x32) -> D[m][n] = sum_c tv[c][n]
// Lane L owns pixel (L%16) and channels 16*(L/16) .. +15 (contiguous K order, not the ISA table order; fine as A is all ones:
// lanes 0-15 K=0..15, lanes 16-31 K=16..31, column N = lane%16).
__global__ __launch_bounds__(256) void tv_chansum_wmma(
    const float* __restrict__ x, float* __restrict__ out) {
  const int lane  = threadIdx.x & 31;
  const int wave  = threadIdx.x >> 5;
  const int p     = lane & 15;          // pixel within the 16-wide tile (WMMA column)
  const int cbase = (lane >> 4) * 16;   // which 16-channel half this lane feeds into B
  const int w0 = blockIdx.x * (WAVES_PER_BLOCK * PIX_PER_WAVE) + wave * PIX_PER_WAVE;
  const int h0 = blockIdx.y * HS;
  const int b  = blockIdx.z;
  const int w  = w0 + p;

  const size_t planeHW = (size_t)Hc * Wc;
  const float* __restrict__ xb = x + (size_t)b * Cc * planeHW;
  float* __restrict__ ob = out + (size_t)b * planeHW;

  // Vertical tap x[h-1] carried in registers while walking down the strip.
  float prev[16];
  if (h0 == 0) {
#pragma unroll
    for (int j = 0; j < 16; ++j) prev[j] = 0.f;  // zero padding at h = -1
  } else {
#pragma unroll
    for (int j = 0; j < 16; ++j)
      prev[j] = xb[(size_t)(cbase + j) * planeHW + (size_t)(h0 - 1) * Wc + w];
  }

  // A = all-ones 16x32 f16 (layout-invariant since every element is 1.0)
  v16h onesA;
#pragma unroll
  for (int j = 0; j < 16; ++j) onesA[j] = (_Float16)1.0f;

  for (int h = h0; h < h0 + HS; ++h) {
    const size_t rowoff = (size_t)h * Wc + w;

    // One coalesced 32-bit load per channel: 32 lanes x 4B = 128B useful/instr.
    float cur[16];
#pragma unroll
    for (int j = 0; j < 16; ++j)
      cur[j] = xb[(size_t)(cbase + j) * planeHW + rowoff];

    // Horizontal tap x[w-1]: neighbor lane's value (DS pipe), except tile edge.
    float left[16];
#pragma unroll
    for (int j = 0; j < 16; ++j)
      left[j] = __shfl_up(cur[j], 1, 32);
    if (p == 0) {
      if (w0 == 0) {
#pragma unroll
        for (int j = 0; j < 16; ++j) left[j] = 0.f;  // zero padding at w = -1
      } else {
#pragma unroll
        for (int j = 0; j < 16; ++j)
          left[j] = xb[(size_t)(cbase + j) * planeHW + rowoff - 1];
      }
    }

    // tv per (channel, pixel), packed into the WMMA B operand (f16 in, f32 acc).
    v16h bm;
#pragma unroll
    for (int j = 0; j < 16; ++j) {
      float t = __builtin_fabsf(prev[j] - cur[j]) + __builtin_fabsf(left[j] - cur[j]);
      bm[j]   = (_Float16)t;
      prev[j] = cur[j];
    }

    // D = ones(16x32) x tv(32x16) + 0  ->  every row of D = channel sums.
    v8f acc = {};
    acc = __builtin_amdgcn_wmma_f32_16x16x32_f16(
        /*neg_a=*/false, onesA, /*neg_b=*/false, bm,
        /*c_mod=*/(short)0, acc, /*reuse_a=*/false, /*reuse_b=*/false);

    // Lanes 0-15 hold D[0][n=lane] in acc[0]: the sum for pixel w0+lane.
    if (lane < 16)
      __builtin_nontemporal_store(acc[0], &ob[rowoff]);
  }
}

extern "C" void kernel_launch(void* const* d_in, const int* in_sizes, int n_in,
                              void* d_out, int out_size, void* d_ws, size_t ws_size,
                              hipStream_t stream) {
  (void)in_sizes; (void)n_in; (void)out_size; (void)d_ws; (void)ws_size;
  const float* x = (const float*)d_in[0];
  float* out = (float*)d_out;
  dim3 grid(Wc / (WAVES_PER_BLOCK * PIX_PER_WAVE), Hc / HS, Bc);
  dim3 block(WAVES_PER_BLOCK * 32);
  hipLaunchKernelGGL(tv_chansum_wmma, grid, block, 0, stream, x, out);
}
